// MPNN_24799141167798
// MI455X (gfx1250) — matrix operand order repaired and checked by reading, not compiled, not run on hardware
//
#include <hip/hip_runtime.h>

typedef __attribute__((ext_vector_type(16))) _Float16 v16h;
typedef __attribute__((ext_vector_type(8)))  _Float16 v8h;
typedef __attribute__((ext_vector_type(8)))  float    v8f;

namespace {
constexpr int kN   = 20000;
constexpr int kE   = 320000;
constexpr int kNF  = 128;
constexpr int kEF  = 32;
constexpr int kH   = 256;
constexpr int kOUT = 128;
constexpr int kL   = 4;
constexpr int kG   = 64;
constexpr int kEC  = 65536;   // edge chunk: multiple of 128; remainder 57856 = 452*128
}

// ---------------------------------------------------------------------------
// CDNA5 async copy helpers (GLOBAL_LOAD_ASYNC_TO_LDS, tracked by ASYNCcnt)
// ---------------------------------------------------------------------------
__device__ __forceinline__ void async_ld_b128(uint32_t lds_off, const void* gaddr) {
  asm volatile("global_load_async_to_lds_b128 %0, %1, off"
               :: "v"(lds_off), "v"(gaddr) : "memory");
}
__device__ __forceinline__ void wait_async_le2() {
  asm volatile("s_wait_asynccnt 0x2" ::: "memory");
}
__device__ __forceinline__ void wait_async_0() {
  asm volatile("s_wait_asynccnt 0x0" ::: "memory");
}

// ---------------------------------------------------------------------------
// Tiled WMMA GEMM:  C[M,Nc] (f32) (+)= A16[gather(rows), K] @ B (via BT[Nc][ldb])
// 256 threads = 8 waves (4 row-groups x 2 col-groups), block tile 128x128,
// K-step 32, double-buffered LDS filled by async-to-LDS DMA, last step peeled
// so the steady-state loop is branch-free: issue -> wait<=2 -> bar -> 8xWMMA -> bar.
// ---------------------------------------------------------------------------
__global__ __launch_bounds__(256)
void k_gemm_wmma(const _Float16* __restrict__ A, int lda,
                 const int* __restrict__ gather, int rowOffset,
                 const _Float16* __restrict__ BT, int ldb,
                 float* __restrict__ C, int M, int Nc, int K, int accumulate) {
  __shared__ __align__(32) _Float16 sA[2][128][32];
  __shared__ __align__(32) _Float16 sB[2][128][32];

  const int t    = threadIdx.x;
  const int wave = t >> 5;        // 0..7
  const int lane = t & 31;
  const int wr   = wave & 3;      // row group: rows [wr*32, wr*32+32)
  const int wc   = wave >> 2;     // col group: cols [wc*64, wc*64+64)
  const int mBase = blockIdx.x * 128;
  const int nBase = blockIdx.y * 128;

  // ---- per-thread staging task: one 16B chunk of A-tile, one of B-tile ----
  const int srow = t >> 1;            // 0..127
  const int tk   = (t & 1) * 8;       // half-offset within K-step (0 or 8)
  int arow = mBase + srow;
  if (arow > M - 1) arow = M - 1;     // clamp: rows >= M never stored, safe dup
  const int aIdx = gather ? gather[rowOffset + arow] : (rowOffset + arow);
  const _Float16* gA = A  + (size_t)aIdx * lda + tk;
  const _Float16* gB = BT + (size_t)(nBase + srow) * ldb + tk;

  const uint32_t ldsA = (uint32_t)(uintptr_t)&sA[0][srow][tk];
  const uint32_t ldsB = (uint32_t)(uintptr_t)&sB[0][srow][tk];
  const uint32_t bufOff = (uint32_t)(128 * 32 * sizeof(_Float16)); // 8KB buffer stride

  v8f acc[2][4];
#pragma unroll
  for (int i = 0; i < 2; ++i)
#pragma unroll
    for (int j = 0; j < 4; ++j)
#pragma unroll
      for (int v = 0; v < 8; ++v) acc[i][j][v] = 0.0f;

  const int rlo   = lane & 15;
  const int khalf = lane >> 4;

  // one K-step of WMMA work from LDS buffer `bufi`
  auto compute = [&](int bufi) {
    v16h afrag[2];
#pragma unroll
    for (int i = 0; i < 2; ++i) {
      const _Float16* ap = &sA[bufi][wr * 32 + i * 16 + rlo][khalf * 16];
      v8h a0 = *(const v8h*)(ap);
      v8h a1 = *(const v8h*)(ap + 8);
#pragma unroll
      for (int q = 0; q < 8; ++q) { afrag[i][q] = a0[q]; afrag[i][8 + q] = a1[q]; }
    }
#pragma unroll
    for (int j = 0; j < 4; ++j) {
      v16h b = *(const v16h*)&sB[bufi][wc * 64 + j * 16 + rlo][khalf * 16];
#pragma unroll
      for (int i = 0; i < 2; ++i) {
        acc[i][j] = __builtin_amdgcn_wmma_f32_16x16x32_f16(
            false, afrag[i], false, b, (short)0, acc[i][j], false, false);
      }
    }
  };

  // preload K-step 0 into buffer 0
  async_ld_b128(ldsA, gA);
  async_ld_b128(ldsB, gB);

  // steady state: all but the last K-step (branch-free body)
  int buf = 0;
  int k0 = 0;
  for (; k0 + 32 < K; k0 += 32) {
    const uint32_t o = (uint32_t)(buf ^ 1) * bufOff;
    async_ld_b128(ldsA + o, gA + k0 + 32);
    async_ld_b128(ldsB + o, gB + k0 + 32);
    wait_async_le2();    // batch for current K-step landed (in-order completion)
    __syncthreads();     // visible to all waves
    compute(buf);
    __syncthreads();     // all waves done reading buf before it is refilled
    buf ^= 1;
  }
  // peeled last K-step
  wait_async_0();
  __syncthreads();
  compute(buf);

  // ---- store: VGPR v -> row (v + 8*(lane>>4)); col = lane&15 ----
  const int rhi = (lane >> 4) * 8;
#pragma unroll
  for (int i = 0; i < 2; ++i)
#pragma unroll
    for (int j = 0; j < 4; ++j)
#pragma unroll
      for (int v = 0; v < 8; ++v) {
        int r = mBase + wr * 32 + i * 16 + v + rhi;
        int c = nBase + wc * 64 + j * 16 + rlo;
        if (r < M) {
          size_t off = (size_t)r * Nc + c;
          C[off] = accumulate ? (C[off] + acc[i][j][v]) : acc[i][j][v];
        }
      }
}

// ---------------------------------------------------------------------------
// Elementwise helpers
// ---------------------------------------------------------------------------
__global__ void k_f32_to_f16(const float* __restrict__ s, _Float16* __restrict__ d, long n) {
  long i = (long)blockIdx.x * blockDim.x + threadIdx.x;
  long st = (long)gridDim.x * blockDim.x;
  for (; i < n; i += st) d[i] = (_Float16)s[i];
}

__global__ void k_transpose_f16(const float* __restrict__ W, _Float16* __restrict__ WT,
                                int K, int Nc) {
  long n = (long)K * Nc;
  long i = (long)blockIdx.x * blockDim.x + threadIdx.x;
  long st = (long)gridDim.x * blockDim.x;
  for (; i < n; i += st) {
    int kk = (int)(i / Nc), nn = (int)(i % Nc);
    WT[(size_t)nn * K + kk] = (_Float16)W[i];
  }
}

__global__ void k_bias_act_f16(const float* __restrict__ T, const float* __restrict__ b,
                               _Float16* __restrict__ O, long M, int Nc, int relu) {
  long n = M * Nc;
  long i = (long)blockIdx.x * blockDim.x + threadIdx.x;
  long st = (long)gridDim.x * blockDim.x;
  for (; i < n; i += st) {
    float v = T[i] + b[i % Nc];
    if (relu) v = fmaxf(v, 0.0f);
    O[i] = (_Float16)v;
  }
}

__global__ void k_bias_f32(const float* __restrict__ T, const float* __restrict__ b,
                           float* __restrict__ O, long M, int Nc) {
  long n = M * Nc;
  long i = (long)blockIdx.x * blockDim.x + threadIdx.x;
  long st = (long)gridDim.x * blockDim.x;
  for (; i < n; i += st) O[i] = T[i] + b[i % Nc];
}

__global__ void k_zero(float* __restrict__ p, long n) {
  long i = (long)blockIdx.x * blockDim.x + threadIdx.x;
  long st = (long)gridDim.x * blockDim.x;
  for (; i < n; i += st) p[i] = 0.0f;
}

__global__ void k_scatter_bias(const float* __restrict__ T, const float* __restrict__ b,
                               const int* __restrict__ dst, int rowOffset, long Mc,
                               float* __restrict__ agg) {
  long n = Mc * kH;
  long i = (long)blockIdx.x * blockDim.x + threadIdx.x;
  long st = (long)gridDim.x * blockDim.x;
  for (; i < n; i += st) {
    long e = i / kH; int c = (int)(i % kH);
    atomicAdd(&agg[(size_t)dst[rowOffset + e] * kH + c], T[i] + b[c]);
  }
}

__global__ void k_pool_sum(const _Float16* __restrict__ h, const int* __restrict__ batch,
                           float* __restrict__ sums, float* __restrict__ cnt) {
  long n = (long)kN * kH;
  long i = (long)blockIdx.x * blockDim.x + threadIdx.x;
  long st = (long)gridDim.x * blockDim.x;
  for (; i < n; i += st) {
    long node = i / kH; int c = (int)(i % kH);
    int g = batch[node];
    atomicAdd(&sums[(size_t)g * kH + c], (float)h[i]);
    if (c == 0) atomicAdd(&cnt[g], 1.0f);
  }
}

__global__ void k_pool_div(const float* __restrict__ sums, const float* __restrict__ cnt,
                           _Float16* __restrict__ pooled) {
  int i = blockIdx.x * blockDim.x + threadIdx.x;
  if (i < kG * kH) {
    int g = i / kH;
    pooled[i] = (_Float16)(sums[i] / fmaxf(cnt[g], 1.0f));
  }
}

// ---------------------------------------------------------------------------
// Host orchestration
// ---------------------------------------------------------------------------
static inline char* carve(char*& p, size_t bytes) {
  char* r = p;
  p += (bytes + 255) & ~((size_t)255);
  return r;
}
static inline int gsblocks(long n) {
  long b = (n + 255) / 256;
  return (int)(b > 32768 ? 32768 : (b < 1 ? 1 : b));
}

extern "C" void kernel_launch(void* const* d_in, const int* in_sizes, int n_in,
                              void* d_out, int out_size, void* d_ws, size_t ws_size,
                              hipStream_t stream) {
  const float* x      = (const float*)d_in[0];
  const float* ea     = (const float*)d_in[1];
  const float* emb_w1 = (const float*)d_in[2];
  const float* emb_b1 = (const float*)d_in[3];
  const float* emb_w2 = (const float*)d_in[4];
  const float* emb_b2 = (const float*)d_in[5];
  const float* mw1    = (const float*)d_in[6];
  const float* mb1    = (const float*)d_in[7];
  const float* mw2    = (const float*)d_in[8];
  const float* mb2    = (const float*)d_in[9];
  const float* uw1    = (const float*)d_in[10];
  const float* ub1    = (const float*)d_in[11];
  const float* uw2    = (const float*)d_in[12];
  const float* ub2    = (const float*)d_in[13];
  const float* hw1    = (const float*)d_in[14];
  const float* hb1    = (const float*)d_in[15];
  const float* hw2    = (const float*)d_in[16];
  const float* hb2    = (const float*)d_in[17];
  const int*   eidx   = (const int*)d_in[18];
  const int*   batch  = (const int*)d_in[19];
  const int*   esrc   = eidx;        // edge_index[0] -> x_j
  const int*   edst   = eidx + kE;   // edge_index[1] -> x_i / scatter target
  float* out = (float*)d_out;

  // ---- workspace carve ----
  char* p = (char*)d_ws;
  _Float16* x16    = (_Float16*)carve(p, (size_t)kN * kNF * 2);
  _Float16* ea16   = (_Float16*)carve(p, (size_t)kE * kEF * 2);
  _Float16* h16    = (_Float16*)carve(p, (size_t)kN * kH * 2);
  float*    tN     = (float*)   carve(p, (size_t)kN * kH * 4);
  _Float16* sN16   = (_Float16*)carve(p, (size_t)kN * kH * 2);
  float*    agg    = (float*)   carve(p, (size_t)kN * kH * 4);
  _Float16* agg16  = (_Float16*)carve(p, (size_t)kN * kH * 2);
  float*    tE     = (float*)   carve(p, (size_t)kEC * kH * 4);
  _Float16* sE16   = (_Float16*)carve(p, (size_t)kEC * kH * 2);
  float*    sums   = (float*)   carve(p, (size_t)kG * kH * 4);
  float*    cnts   = (float*)   carve(p, (size_t)kG * 4);
  _Float16* pool16 = (_Float16*)carve(p, (size_t)kG * kH * 2);
  float*    tG     = (float*)   carve(p, (size_t)kG * kH * 4);
  _Float16* sG16   = (_Float16*)carve(p, (size_t)kG * kH * 2);
  _Float16* wEmb1T = (_Float16*)carve(p, (size_t)kNF * kH * 2);
  _Float16* wEmb2T = (_Float16*)carve(p, (size_t)kH * kH * 2);
  _Float16* wM1T   = (_Float16*)carve(p, (size_t)kL * (2 * kH + kEF) * kH * 2);
  _Float16* wM2T   = (_Float16*)carve(p, (size_t)kL * kH * kH * 2);
  _Float16* wU1T   = (_Float16*)carve(p, (size_t)kL * 2 * kH * kH * 2);
  _Float16* wU2T   = (_Float16*)carve(p, (size_t)kL * kH * kH * 2);
  _Float16* wH1T   = (_Float16*)carve(p, (size_t)kH * kH * 2);
  _Float16* wH2T   = (_Float16*)carve(p, (size_t)kH * kOUT * 2);

  auto gemm = [&](const _Float16* A, int lda, const int* gather, int rowOffset,
                  const _Float16* BT, int ldb, float* C, int M, int Nc, int K, int acc) {
    dim3 g((M + 127) / 128, (Nc + 127) / 128), b(256);
    k_gemm_wmma<<<g, b, 0, stream>>>(A, lda, gather, rowOffset, BT, ldb, C, M, Nc, K, acc);
  };

  // ---- convert inputs / transpose weights to f16 (BT layout, K contiguous) ----
  k_f32_to_f16<<<gsblocks((long)kN * kNF), 256, 0, stream>>>(x, x16, (long)kN * kNF);
  k_f32_to_f16<<<gsblocks((long)kE * kEF), 256, 0, stream>>>(ea, ea16, (long)kE * kEF);
  k_transpose_f16<<<gsblocks((long)kNF * kH), 256, 0, stream>>>(emb_w1, wEmb1T, kNF, kH);
  k_transpose_f16<<<gsblocks((long)kH * kH), 256, 0, stream>>>(emb_w2, wEmb2T, kH, kH);
  k_transpose_f16<<<gsblocks((long)kH * kH), 256, 0, stream>>>(hw1, wH1T, kH, kH);
  k_transpose_f16<<<gsblocks((long)kH * kOUT), 256, 0, stream>>>(hw2, wH2T, kH, kOUT);
  const int KM = 2 * kH + kEF;   // 544
  const int KU = 2 * kH;         // 512
  for (int l = 0; l < kL; ++l) {
    k_transpose_f16<<<gsblocks((long)KM * kH), 256, 0, stream>>>(
        mw1 + (size_t)l * KM * kH, wM1T + (size_t)l * KM * kH, KM, kH);
    k_transpose_f16<<<gsblocks((long)kH * kH), 256, 0, stream>>>(
        mw2 + (size_t)l * kH * kH, wM2T + (size_t)l * kH * kH, kH, kH);
    k_transpose_f16<<<gsblocks((long)KU * kH), 256, 0, stream>>>(
        uw1 + (size_t)l * KU * kH, wU1T + (size_t)l * KU * kH, KU, kH);
    k_transpose_f16<<<gsblocks((long)kH * kH), 256, 0, stream>>>(
        uw2 + (size_t)l * kH * kH, wU2T + (size_t)l * kH * kH, kH, kH);
  }

  // ---- embedder MLP: h = (relu(x@W1+b1))@W2 + b2 ----
  gemm(x16, kNF, nullptr, 0, wEmb1T, kNF, tN, kN, kH, kNF, 0);
  k_bias_act_f16<<<gsblocks((long)kN * kH), 256, 0, stream>>>(tN, emb_b1, sN16, kN, kH, 1);
  gemm(sN16, kH, nullptr, 0, wEmb2T, kH, tN, kN, kH, kH, 0);
  k_bias_act_f16<<<gsblocks((long)kN * kH), 256, 0, stream>>>(tN, emb_b2, h16, kN, kH, 0);

  // ---- L message-passing layers ----
  for (int l = 0; l < kL; ++l) {
    const _Float16* m1T = wM1T + (size_t)l * KM * kH;  // column stride = 544
    const _Float16* m2T = wM2T + (size_t)l * kH * kH;
    const _Float16* u1T = wU1T + (size_t)l * KU * kH;  // column stride = 512
    const _Float16* u2T = wU2T + (size_t)l * kH * kH;

    k_zero<<<gsblocks((long)kN * kH), 256, 0, stream>>>(agg, (long)kN * kH);

    for (int cs = 0; cs < kE; cs += kEC) {
      int Mc = (kE - cs < kEC) ? (kE - cs) : kEC;
      // t1 = h[dst]@W[0:256] + h[src]@W[256:512] + ea@W[512:544]   (K-split of concat)
      gemm(h16, kH, edst, cs, m1T,        KM, tE, Mc, kH, kH,  0);
      gemm(h16, kH, esrc, cs, m1T + kH,   KM, tE, Mc, kH, kH,  1);
      gemm(ea16, kEF, nullptr, cs, m1T + 2 * kH, KM, tE, Mc, kH, kEF, 1);
      k_bias_act_f16<<<gsblocks((long)Mc * kH), 256, 0, stream>>>(tE, mb1 + l * kH, sE16, Mc, kH, 1);
      // m = s1 @ W2 (+b2 folded into scatter), then agg[dst] += m
      gemm(sE16, kH, nullptr, 0, m2T, kH, tE, Mc, kH, kH, 0);
      k_scatter_bias<<<gsblocks((long)Mc * kH), 256, 0, stream>>>(tE, mb2 + l * kH, edst, cs, Mc, agg);
    }

    // update net: h = (relu(h@U[0:256] + agg@U[256:512] + b1))@U2 + b2
    k_f32_to_f16<<<gsblocks((long)kN * kH), 256, 0, stream>>>(agg, agg16, (long)kN * kH);
    gemm(h16,  kH, nullptr, 0, u1T,      KU, tN, kN, kH, kH, 0);
    gemm(agg16, kH, nullptr, 0, u1T + kH, KU, tN, kN, kH, kH, 1);
    k_bias_act_f16<<<gsblocks((long)kN * kH), 256, 0, stream>>>(tN, ub1 + l * kH, sN16, kN, kH, 1);
    gemm(sN16, kH, nullptr, 0, u2T, kH, tN, kN, kH, kH, 0);
    k_bias_act_f16<<<gsblocks((long)kN * kH), 256, 0, stream>>>(tN, ub2 + l * kH, h16, kN, kH, 0);
  }

  // ---- global mean pool ----
  k_zero<<<gsblocks((long)kG * kH), 256, 0, stream>>>(sums, (long)kG * kH);
  k_zero<<<1, 256, 0, stream>>>(cnts, (long)kG);
  k_pool_sum<<<gsblocks((long)kN * kH), 256, 0, stream>>>(h16, batch, sums, cnts);
  k_pool_div<<<(kG * kH + 255) / 256, 256, 0, stream>>>(sums, cnts, pool16);

  // ---- head MLP -> d_out [G, OUT] f32 ----
  gemm(pool16, kH, nullptr, 0, wH1T, kH, tG, kG, kH, kH, 0);
  k_bias_act_f16<<<gsblocks((long)kG * kH), 256, 0, stream>>>(tG, hb1, sG16, kG, kH, 1);
  gemm(sG16, kH, nullptr, 0, wH2T, kH, tG, kG, kOUT, kH, 0);
  k_bias_f32<<<gsblocks((long)kG * kOUT), 256, 0, stream>>>(tG, hb2, out, kG, kOUT);
}
